// MultiHeadSelfAttn_10256381903452
// MI455X (gfx1250) — matrix operand order repaired and checked by reading, not compiled, not run on hardware
//
#include <hip/hip_runtime.h>
#include <hip/hip_bf16.h>

typedef __attribute__((ext_vector_type(8)))  float          v8f;
typedef __attribute__((ext_vector_type(16))) __bf16         v16bf;
typedef __attribute__((ext_vector_type(8)))  unsigned short u16x8;
typedef __attribute__((ext_vector_type(16))) unsigned short u16x16;

#define NB     4
#define SLEN   2048
#define DMODEL 1024
#define NH     16
#define DK     64

// ---------- helpers ----------
static __device__ __forceinline__ unsigned short f2bf(float f) {
  unsigned int u = __builtin_bit_cast(unsigned int, f);
  u += 0x7FFFu + ((u >> 16) & 1u);          // round-to-nearest-even
  return (unsigned short)(u >> 16);
}

// Build a 16-element bf16 fragment from two contiguous 16-byte chunks.
static __device__ __forceinline__ v16bf ld_frag(const unsigned short* p0,
                                                const unsigned short* p1) {
  u16x8 lo = *(const u16x8*)p0;
  u16x8 hi = *(const u16x8*)p1;
  u16x16 u = __builtin_shufflevector(lo, hi, 0,1,2,3,4,5,6,7,8,9,10,11,12,13,14,15);
  return __builtin_bit_cast(v16bf, u);
}

static __device__ __forceinline__ v8f wmma_bf16(v16bf a, v16bf b, v8f c) {
  return __builtin_amdgcn_wmma_f32_16x16x32_bf16(false, a, false, b,
                                                 (short)0, c, false, false);
}

// CDNA5 async copy: 16B per lane, global -> LDS, tracked by ASYNCcnt.
static __device__ __forceinline__ void async_cp16(unsigned int lds_off,
                                                  const void* gptr) {
  unsigned long long ga = (unsigned long long)gptr;
  asm volatile("global_load_async_to_lds_b128 %0, %1, off"
               :: "v"(lds_off), "v"(ga) : "memory");
}

// ---------- fp32 -> bf16 convert ----------
__global__ void k_f32_to_bf16(const float* __restrict__ in,
                              unsigned short* __restrict__ out, int n) {
  int i = blockIdx.x * blockDim.x + threadIdx.x;
  if (i < n) out[i] = f2bf(in[i]);
}

// ---------- GEMM: C[M][N] = A[M][K](bf16) * B[N][K]^T(bf16), f32 out ----------
// 8 waves / block; block tile 64x256; wave tile 32x64 (2 A-frags x 4 B-frags).
// A/B tiles staged in LDS via global_load_async_to_lds_b128, double-buffered,
// pipeline unrolled 2x so both buffer indices are compile-time constants.
__global__ __launch_bounds__(256) void k_gemm_bf16_nt(
    const unsigned short* __restrict__ A, const unsigned short* __restrict__ Bw,
    float* __restrict__ C, int M, int N, int K) {
  __shared__ unsigned short sA[2][64 * 32];    //  2 x 4 KB
  __shared__ unsigned short sB[2][256 * 32];   //  2 x 16 KB

  const int tid  = threadIdx.x;
  const int lane = tid & 31;
  const int wave = tid >> 5;
  const int l15 = lane & 15, lhi = lane >> 4;
  const int wm = wave >> 2, wn = wave & 3;
  const int rb = blockIdx.y * 64;
  const int cb = blockIdx.x * 256;

  // per-thread copy assignment: 16B granules; row = tid/4, kchunk = (tid%4)*8
  const int crow = tid >> 2, ckc = (tid & 3) * 8;
  const unsigned short* gA = A + (size_t)(rb + crow) * K + ckc;
  const unsigned short* gB0 = Bw + (size_t)(cb + crow) * K + ckc;

  unsigned int ldsA[2], ldsB[2][4];
#pragma unroll
  for (int f = 0; f < 2; ++f) {
    ldsA[f] = (unsigned int)(size_t)&sA[f][crow * 32 + ckc];
#pragma unroll
    for (int it = 0; it < 4; ++it)
      ldsB[f][it] = (unsigned int)(size_t)&sB[f][(crow + 64 * it) * 32 + ckc];
  }

  v8f acc[2][4] = {};
  const int aoff = (wm * 32 + l15) * 32 + lhi * 8;   // A frag base (elem)
  const int boff = (wn * 64 + l15) * 32 + lhi * 8;   // B frag base (elem)

  // f must be a compile-time constant at every call site (no movrels).
  auto issue = [&](int f, int kt) {
    async_cp16(ldsA[f], gA + kt);
#pragma unroll
    for (int it = 0; it < 4; ++it)
      async_cp16(ldsB[f][it], gB0 + (size_t)(64 * it) * K + kt);
  };
  auto compute = [&](const unsigned short* pa, const unsigned short* pb) {
    v16bf a0 = ld_frag(pa + aoff,           pa + aoff + 16);
    v16bf a1 = ld_frag(pa + aoff + 16 * 32, pa + aoff + 16 * 32 + 16);
#pragma unroll
    for (int j = 0; j < 4; j++) {
      const unsigned short* pj = pb + boff + j * 16 * 32;
      v16bf b = ld_frag(pj, pj + 16);
      acc[0][j] = wmma_bf16(a0, b, acc[0][j]);
      acc[1][j] = wmma_bf16(a1, b, acc[1][j]);
    }
  };

  issue(0, 0);
  for (int kt = 0; kt < K; kt += 64) {               // K % 64 == 0
    asm volatile("s_wait_asynccnt 0x0" ::: "memory");
    __syncthreads();
    issue(1, kt + 32);                                // always valid
    compute(&sA[0][0], &sB[0][0]);

    asm volatile("s_wait_asynccnt 0x0" ::: "memory");
    __syncthreads();
    if (kt + 64 < K) issue(0, kt + 64);
    compute(&sA[1][0], &sB[1][0]);
  }

#pragma unroll
  for (int i = 0; i < 2; i++)
#pragma unroll
    for (int j = 0; j < 4; j++)
#pragma unroll
      for (int r = 0; r < 8; r++) {
        int row = rb + wm * 32 + i * 16 + r + 8 * lhi;
        int col = cb + wn * 64 + j * 16 + l15;
        C[(size_t)row * N + col] = acc[i][j][r];
      }
}

// ---------- RoPE + convert: fp32 (B,S,D) -> bf16 (B,H,S,dk) ----------
__global__ void k_rope_cvt(const float* __restrict__ Xf,
                           const int* __restrict__ pos,
                           unsigned short* __restrict__ Xb) {
  int idx = blockIdx.x * blockDim.x + threadIdx.x;  // NB*SLEN*512 pairs
  int d2 = idx & 511;
  int s  = (idx >> 9) & (SLEN - 1);
  int b  = idx >> 20;
  int h = d2 >> 5, i = d2 & 31;
  float p   = (float)pos[b * SLEN + s];
  float inv = __expf(-(float)(2 * i) * (9.210340371976184f / 64.f)); // 1/theta^(2i/dk)
  float ang = p * inv;
  float c = __cosf(ang), sn = __sinf(ang);
  const float* src = Xf + ((size_t)b * SLEN + s) * DMODEL + h * DK + 2 * i;
  float e = src[0], o = src[1];
  unsigned short* dst =
      Xb + ((size_t)(b * NH + h) * SLEN + s) * DK + 2 * i;
  dst[0] = f2bf(e * c - o * sn);
  dst[1] = f2bf(e * sn + o * c);
}

// ---------- V: fp32 (B,S,D) -> bf16 transposed (B,H,dk,S) ----------
__global__ void k_cvt_v_t(const float* __restrict__ Vf,
                          unsigned short* __restrict__ Vt) {
  int idx = blockIdx.x * blockDim.x + threadIdx.x;  // NB*SLEN*DMODEL
  int d = idx & (DMODEL - 1);
  int s = (idx >> 10) & (SLEN - 1);
  int b = idx >> 21;
  int h = d >> 6, dv = d & 63;
  Vt[((size_t)(b * NH + h) * DK + dv) * SLEN + s] = f2bf(Vf[idx]);
}

// ---------- causal flash attention, one 16-row q-tile per wave ----------
__global__ __launch_bounds__(256) void k_flash_attn(
    const unsigned short* __restrict__ Qb, const unsigned short* __restrict__ Kb,
    const unsigned short* __restrict__ Vt, unsigned short* __restrict__ Ob) {
  __shared__ unsigned short psh[8][16 * 32];   // per-wave P tile (16q x 32k bf16)
  const int lane = threadIdx.x & 31;
  const int wave = threadIdx.x >> 5;
  const int l15 = lane & 15, lhi = lane >> 4;
  const int h = blockIdx.y, b = blockIdx.z;
  const int q0 = (blockIdx.x * 8 + wave) * 16;

  const unsigned short* Qh = Qb + (size_t)(b * NH + h) * SLEN * DK;
  const unsigned short* Kh = Kb + (size_t)(b * NH + h) * SLEN * DK;
  const unsigned short* Vh = Vt + (size_t)(b * NH + h) * DK * SLEN;

  // Q A-fragments (row = token, K-dim = dk, 2 frags cover dk=64)
  v16bf qa0, qa1;
  {
    const unsigned short* qr = Qh + (size_t)(q0 + l15) * DK + lhi * 16;
    qa0 = ld_frag(qr, qr + 8);
    qa1 = ld_frag(qr + 32, qr + 40);
  }

  v8f o[4] = {};
  float m[8], l[8];
#pragma unroll
  for (int r = 0; r < 8; r++) { m[r] = -__builtin_inff(); l[r] = 0.f; }

  const int jbmax = (q0 + 15) >> 5;
  for (int jb = 0; jb <= jbmax; ++jb) {
    const int j0 = jb << 5;

    // scores for 32 keys: two 16x16 tiles, contraction over dk=64 (2 WMMA each)
    v8f s0 = {}, s1 = {};
    {
      const unsigned short* kr = Kh + (size_t)(j0 + l15) * DK + lhi * 16;
      s0 = wmma_bf16(qa0, ld_frag(kr, kr + 8), s0);
      s0 = wmma_bf16(qa1, ld_frag(kr + 32, kr + 40), s0);
    }
    {
      const unsigned short* kr = Kh + (size_t)(j0 + 16 + l15) * DK + lhi * 16;
      s1 = wmma_bf16(qa0, ld_frag(kr, kr + 8), s1);
      s1 = wmma_bf16(qa1, ld_frag(kr + 32, kr + 40), s1);
    }

    // scale + causal mask
    const int qrb = q0 + 8 * lhi;
    float bm[8], al[8], ps[8];
#pragma unroll
    for (int r = 0; r < 8; r++) {
      int qi = qrb + r;
      float a = (j0 + l15      <= qi) ? s0[r] * 0.125f : -__builtin_inff();
      float c = (j0 + 16 + l15 <= qi) ? s1[r] * 0.125f : -__builtin_inff();
      s0[r] = a; s1[r] = c;
      bm[r] = fmaxf(a, c);
    }
    // row-max across the 16 lanes holding one row (xor 1,2,4,8 stays in half)
#pragma unroll
    for (int r = 0; r < 8; r++) {
      bm[r] = fmaxf(bm[r], __shfl_xor(bm[r], 1));
      bm[r] = fmaxf(bm[r], __shfl_xor(bm[r], 2));
      bm[r] = fmaxf(bm[r], __shfl_xor(bm[r], 4));
      bm[r] = fmaxf(bm[r], __shfl_xor(bm[r], 8));
    }
#pragma unroll
    for (int r = 0; r < 8; r++) {
      float nm = fmaxf(m[r], bm[r]);
      al[r] = __expf(m[r] - nm);
      m[r] = nm;
      float p0 = __expf(s0[r] - nm);
      float p1 = __expf(s1[r] - nm);
      s0[r] = p0; s1[r] = p1;
      ps[r] = p0 + p1;
    }
#pragma unroll
    for (int r = 0; r < 8; r++) {
      ps[r] += __shfl_xor(ps[r], 1);
      ps[r] += __shfl_xor(ps[r], 2);
      ps[r] += __shfl_xor(ps[r], 4);
      ps[r] += __shfl_xor(ps[r], 8);
      l[r] = l[r] * al[r] + ps[r];
    }
#pragma unroll
    for (int f = 0; f < 4; f++)
#pragma unroll
      for (int r = 0; r < 8; r++) o[f][r] *= al[r];

    // C-frag -> A-frag layout change for P via LDS (wave-private, 1 KB)
    unsigned short* pw = psh[wave];
#pragma unroll
    for (int r = 0; r < 8; r++) {
      pw[(r + 8 * lhi) * 32 + l15]      = f2bf(s0[r]);
      pw[(r + 8 * lhi) * 32 + 16 + l15] = f2bf(s1[r]);
    }
    asm volatile("s_wait_dscnt 0x0" ::: "memory");  // same-wave LDS RAW
    const unsigned short* pr = pw + l15 * 32 + lhi * 16;
    v16bf pa = ld_frag(pr, pr + 8);

    // O += P(16x32) x V(32x64): 4 d-tiles, V^T layout makes K contiguous
#pragma unroll
    for (int f = 0; f < 4; f++) {
      const unsigned short* vr = Vh + (size_t)(f * 16 + l15) * SLEN + j0 + lhi * 16;
      o[f] = wmma_bf16(pa, ld_frag(vr, vr + 8), o[f]);
    }
  }

  // normalize and write bf16 attention output in (B,S,D) for the final GEMM
#pragma unroll
  for (int r = 0; r < 8; r++) l[r] = 1.f / l[r];
  unsigned short* ob = Ob + (size_t)b * SLEN * DMODEL + h * DK;
#pragma unroll
  for (int f = 0; f < 4; f++)
#pragma unroll
    for (int r = 0; r < 8; r++) {
      int qq = q0 + r + 8 * lhi;
      ob[(size_t)qq * DMODEL + f * 16 + l15] = f2bf(o[f][r] * l[r]);
    }
}

// ---------- launcher ----------
extern "C" void kernel_launch(void* const* d_in, const int* in_sizes, int n_in,
                              void* d_out, int out_size, void* d_ws, size_t ws_size,
                              hipStream_t stream) {
  (void)in_sizes; (void)n_in; (void)out_size; (void)ws_size;
  const float* x  = (const float*)d_in[0];
  const int*   tp = (const int*)d_in[1];
  const float* wq = (const float*)d_in[2];
  const float* wk = (const float*)d_in[3];
  const float* wv = (const float*)d_in[4];
  const float* wo = (const float*)d_in[5];

  char* w = (char*)d_ws;
  unsigned short* Xb  = (unsigned short*)(w);                       //  0..16 MiB
  unsigned short* Wqb = (unsigned short*)(w + ((size_t)16 << 20));  //  2 MiB each
  unsigned short* Wkb = (unsigned short*)(w + ((size_t)18 << 20));
  unsigned short* Wvb = (unsigned short*)(w + ((size_t)20 << 20));
  unsigned short* Wob = (unsigned short*)(w + ((size_t)22 << 20));
  float* Qf = (float*)(w + ((size_t)24 << 20));                     // 32 MiB
  float* Kf = (float*)(w + ((size_t)56 << 20));
  float* Vf = (float*)(w + ((size_t)88 << 20));
  unsigned short* QB = (unsigned short*)(w + ((size_t)120 << 20));  // 16 MiB
  unsigned short* KB = (unsigned short*)(w + ((size_t)136 << 20));
  unsigned short* VT = (unsigned short*)(w + ((size_t)152 << 20));
  unsigned short* OB = (unsigned short*)(w + ((size_t)24 << 20));   // reuse Qf

  const int M = NB * SLEN;               // 8192
  const int nX = M * DMODEL;             // 8.39 M elements
  const int nW = DMODEL * DMODEL;

  k_f32_to_bf16<<<dim3((nX + 255) / 256), 256, 0, stream>>>(x, Xb, nX);
  k_f32_to_bf16<<<dim3((nW + 255) / 256), 256, 0, stream>>>(wq, Wqb, nW);
  k_f32_to_bf16<<<dim3((nW + 255) / 256), 256, 0, stream>>>(wk, Wkb, nW);
  k_f32_to_bf16<<<dim3((nW + 255) / 256), 256, 0, stream>>>(wv, Wvb, nW);
  k_f32_to_bf16<<<dim3((nW + 255) / 256), 256, 0, stream>>>(wo, Wob, nW);

  dim3 gg(DMODEL / 256, M / 64);         // (4, 128), 256 threads
  k_gemm_bf16_nt<<<gg, 256, 0, stream>>>(Xb, Wqb, Qf, M, DMODEL, DMODEL);
  k_gemm_bf16_nt<<<gg, 256, 0, stream>>>(Xb, Wkb, Kf, M, DMODEL, DMODEL);
  k_gemm_bf16_nt<<<gg, 256, 0, stream>>>(Xb, Wvb, Vf, M, DMODEL, DMODEL);

  const int nR = NB * SLEN * 512;
  k_rope_cvt<<<dim3(nR / 256), 256, 0, stream>>>(Qf, tp, QB);
  k_rope_cvt<<<dim3(nR / 256), 256, 0, stream>>>(Kf, tp, KB);
  k_cvt_v_t<<<dim3(nX / 256), 256, 0, stream>>>(Vf, VT);

  k_flash_attn<<<dim3(SLEN / 128, NH, NB), 256, 0, stream>>>(QB, KB, VT, OB);

  k_gemm_bf16_nt<<<gg, 256, 0, stream>>>(OB, Wob, (float*)d_out, M, DMODEL, DMODEL);
}
